// DeepseekV32DecoderLayer_2259152798606
// MI455X (gfx1250) — compile-verified, matrix operand order checked by reading, not run on hardware
//
#include <hip/hip_runtime.h>
#include <stdint.h>

namespace {

constexpr int T_ = 1024;
constexpr int D_ = 2048;
constexpr int H_ = 16;
constexpr int NOPE_ = 128, ROPE_ = 64, VD_ = 128;
constexpr int QKD_ = NOPE_ + ROPE_;          // 192
constexpr int QL_ = 1536, KVL_ = 512;
constexpr int QKV_N = QL_ + KVL_ + ROPE_;    // 2112
constexpr int IH_ = 32, ID_ = 128, TOPK_ = 512;
constexpr int E_ = 16, TK_ = 4, I_ = 1408, ISH_ = 2816;

typedef __attribute__((ext_vector_type(16))) __bf16 bf16x16;
typedef __attribute__((ext_vector_type(8)))  __bf16 bf16x8;
typedef __attribute__((ext_vector_type(4)))  __bf16 bf16x4;
typedef __attribute__((ext_vector_type(8)))  float  f32x8;

union Frag { bf16x16 v; bf16x8 h[2]; };

__device__ __forceinline__ f32x8 wmma_bf16(const Frag& a, const Frag& b, f32x8 c) {
  // V_WMMA_F32_16X16X32_BF16: D = A(16x32) * B(32x16) + C(16x16 f32)
  return __builtin_amdgcn_wmma_f32_16x16x32_bf16(false, a.v, false, b.v, (short)0, c, false, false);
}

__device__ __forceinline__ bf16x4 cvt4(float4 v){
  bf16x4 o; o[0]=(__bf16)v.x; o[1]=(__bf16)v.y; o[2]=(__bf16)v.z; o[3]=(__bf16)v.w;
  return o;
}

__device__ __forceinline__ float sigmoidf_(float x){ return 1.f/(1.f+__expf(-x)); }
__device__ __forceinline__ float siluf_(float x){ return x*sigmoidf_(x); }
// inv_freq[j] = 10000^(-j/32), j in [0,32)
__device__ __forceinline__ float rope_freq(int j){ return __expf(-(float)j * (9.210340371976184f/32.0f)); }
__device__ __forceinline__ unsigned fkey(float f){
  unsigned u=__float_as_uint(f); return (u&0x80000000u)? ~u : (u|0x80000000u);
}

// ---------------------------------------------------------------------------
// Generic tiled GEMM: C[MxN] = A[MxK] * B[KxN], fp32 in/out, bf16 WMMA inside.
// BT=false: B row-major KxN.  BT=true: B is row-major NxK (i.e. logical B^T).
// Requirements (hold at all call sites): K%32==0, N%4==0, lda%4==0, ldb%4==0,
// base pointers 16B aligned.
// ---------------------------------------------------------------------------
template<bool BT>
__global__ __launch_bounds__(256) void k_gemm(const float* __restrict__ A,
                                              const float* __restrict__ B,
                                              float* __restrict__ C,
                                              int M, int N, int K,
                                              int lda, int ldb, int ldc)
{
  constexpr int BM=128, BN=128, BK=32, LS=40; // LS: LDS row stride (bf16), 80B = 5x16B
  __shared__ __align__(16) __bf16 sA[BM*LS];
  __shared__ __align__(16) __bf16 sB[BN*LS];
  const int tid  = threadIdx.x;
  const int wave = tid>>5, lane = tid&31;
  const int wM = wave & 3, wN = wave >> 2;     // 4 x 2 waves
  const int half = lane>>4, lr = lane&15;
  const int tM = blockIdx.y*BM, tN = blockIdx.x*BN;

  const f32x8 z8 = {0.f,0.f,0.f,0.f,0.f,0.f,0.f,0.f};
  f32x8 acc[2][4];
#pragma unroll
  for (int i=0;i<2;++i)
#pragma unroll
    for (int j=0;j<4;++j) acc[i][j]=z8;

  for (int k0=0;k0<K;k0+=BK) {
    const bool pf = (k0+BK)<K;
    // ---- A tile: 128x32, float4 global loads -> packed bf16x4 LDS stores ----
#pragma unroll
    for (int i=tid;i<(BM*BK)/4;i+=256){
      int r=i>>3, c=(i&7)*4; int gr=tM+r;
      float4 v = {0.f,0.f,0.f,0.f};
      if (gr<M) {
        const float* ap = &A[(size_t)gr*lda + k0 + c];
        v = *(const float4*)ap;
        if (pf) __builtin_prefetch(ap + BK, 0, 1);   // global_prefetch next K tile
      }
      *(bf16x4*)&sA[r*LS+c] = cvt4(v);
    }
    // ---- B tile ----
    if (BT) {
#pragma unroll
      for (int i=tid;i<(BN*BK)/4;i+=256){
        int n=i>>3, c=(i&7)*4; int gn=tN+n;
        float4 v = {0.f,0.f,0.f,0.f};
        if (gn<N) {
          const float* bp = &B[(size_t)gn*ldb + k0 + c];
          v = *(const float4*)bp;
          if (pf) __builtin_prefetch(bp + BK, 0, 1);
        }
        *(bf16x4*)&sB[n*LS+c] = cvt4(v);
      }
    } else {
#pragma unroll
      for (int i=tid;i<(BN*BK)/4;i+=256){
        int k=i>>5, n4=(i&31)*4; int gn=tN+n4;
        float4 v = {0.f,0.f,0.f,0.f};
        if (gn<N) {             // N%4==0 -> whole float4 in/out uniformly
          const float* bp = &B[(size_t)(k0+k)*ldb + gn];
          v = *(const float4*)bp;
          if (pf) __builtin_prefetch(bp + (size_t)BK*ldb, 0, 1);
        }
        sB[(n4+0)*LS+k]=(__bf16)v.x; sB[(n4+1)*LS+k]=(__bf16)v.y;
        sB[(n4+2)*LS+k]=(__bf16)v.z; sB[(n4+3)*LS+k]=(__bf16)v.w;
      }
    }
    __syncthreads();
    Frag a[2], b[4];
#pragma unroll
    for (int mt=0;mt<2;++mt){
      int r = wM*32 + mt*16 + lr;
      a[mt].h[0] = *(const bf16x8*)&sA[r*LS + half*8];
      a[mt].h[1] = *(const bf16x8*)&sA[r*LS + 16 + half*8];
    }
#pragma unroll
    for (int nt=0;nt<4;++nt){
      int c = wN*64 + nt*16 + lr;
      b[nt].h[0] = *(const bf16x8*)&sB[c*LS + half*8];
      b[nt].h[1] = *(const bf16x8*)&sB[c*LS + 16 + half*8];
    }
#pragma unroll
    for (int mt=0;mt<2;++mt)
#pragma unroll
      for (int nt=0;nt<4;++nt)
        acc[mt][nt] = wmma_bf16(a[mt], b[nt], acc[mt][nt]);
    __syncthreads();
  }
#pragma unroll
  for (int mt=0;mt<2;++mt)
#pragma unroll
    for (int nt=0;nt<4;++nt)
#pragma unroll
      for (int j=0;j<8;++j){
        int row = tM + wM*32 + mt*16 + half*8 + j;
        int col = tN + wN*64 + nt*16 + lr;
        if (row<M && col<N) C[(size_t)row*ldc + col] = acc[mt][nt][j];
      }
}

// ---------------------------------------------------------------------------
// Grouped (per-expert) GEMM over expert-contiguous row segments.
// GA: gather A rows via tokmap; SC: scatter-add C rows via tokmap (atomic).
// ---------------------------------------------------------------------------
template<bool GA, bool SC>
__global__ __launch_bounds__(256) void k_gemm_grp(const float* __restrict__ A,
                                                  const float* __restrict__ Bbase,
                                                  float* __restrict__ C,
                                                  const int* __restrict__ tokmap,
                                                  const int* __restrict__ seg,
                                                  int N, int K, int lda,
                                                  long long strideB, int ldc)
{
  constexpr int BM=128, BN=128, BK=32, LS=40;
  const int e = blockIdx.z;
  const int seg0 = seg[e];
  const int cnt  = seg[e+1]-seg0;
  const int tM = blockIdx.y*BM;
  if (tM >= cnt) return;
  const int tN = blockIdx.x*BN;
  const float* B = Bbase + (size_t)e*(size_t)strideB;

  __shared__ __align__(16) __bf16 sA[BM*LS];
  __shared__ __align__(16) __bf16 sB[BN*LS];
  const int tid  = threadIdx.x;
  const int wave = tid>>5, lane = tid&31;
  const int wM = wave & 3, wN = wave >> 2;
  const int half = lane>>4, lr = lane&15;

  const f32x8 z8 = {0.f,0.f,0.f,0.f,0.f,0.f,0.f,0.f};
  f32x8 acc[2][4];
#pragma unroll
  for (int i=0;i<2;++i)
#pragma unroll
    for (int j=0;j<4;++j) acc[i][j]=z8;

  for (int k0=0;k0<K;k0+=BK) {
    const bool pf = (k0+BK)<K;
#pragma unroll
    for (int i=tid;i<(BM*BK)/4;i+=256){
      int r=i>>3, c=(i&7)*4; int rl=tM+r;
      float4 v = {0.f,0.f,0.f,0.f};
      if (rl<cnt){
        int srow = GA ? tokmap[seg0+rl] : (seg0+rl);
        const float* ap = &A[(size_t)srow*lda + k0 + c];
        v = *(const float4*)ap;
        if (pf) __builtin_prefetch(ap + BK, 0, 1);
      }
      *(bf16x4*)&sA[r*LS+c] = cvt4(v);
    }
#pragma unroll
    for (int i=tid;i<(BN*BK)/4;i+=256){
      int k=i>>5, n4=(i&31)*4; int gn=tN+n4;
      float4 v = {0.f,0.f,0.f,0.f};
      if (gn<N) {
        const float* bp = &B[(size_t)(k0+k)*N + gn];
        v = *(const float4*)bp;
        if (pf) __builtin_prefetch(bp + (size_t)BK*N, 0, 1);
      }
      sB[(n4+0)*LS+k]=(__bf16)v.x; sB[(n4+1)*LS+k]=(__bf16)v.y;
      sB[(n4+2)*LS+k]=(__bf16)v.z; sB[(n4+3)*LS+k]=(__bf16)v.w;
    }
    __syncthreads();
    Frag a[2], b[4];
#pragma unroll
    for (int mt=0;mt<2;++mt){
      int r = wM*32 + mt*16 + lr;
      a[mt].h[0] = *(const bf16x8*)&sA[r*LS + half*8];
      a[mt].h[1] = *(const bf16x8*)&sA[r*LS + 16 + half*8];
    }
#pragma unroll
    for (int nt=0;nt<4;++nt){
      int c = wN*64 + nt*16 + lr;
      b[nt].h[0] = *(const bf16x8*)&sB[c*LS + half*8];
      b[nt].h[1] = *(const bf16x8*)&sB[c*LS + 16 + half*8];
    }
#pragma unroll
    for (int mt=0;mt<2;++mt)
#pragma unroll
      for (int nt=0;nt<4;++nt)
        acc[mt][nt] = wmma_bf16(a[mt], b[nt], acc[mt][nt]);
    __syncthreads();
  }
#pragma unroll
  for (int mt=0;mt<2;++mt)
#pragma unroll
    for (int nt=0;nt<4;++nt)
#pragma unroll
      for (int j=0;j<8;++j){
        int rl  = tM + wM*32 + mt*16 + half*8 + j;
        int col = tN + wN*64 + nt*16 + lr;
        if (rl<cnt && col<N){
          float v = acc[mt][nt][j];
          if (SC) atomicAdd(&C[(size_t)tokmap[seg0+rl]*ldc + col], v);
          else    C[(size_t)(seg0+rl)*ldc + col] = v;
        }
      }
}

// ---------------------------------------------------------------------------
// Fused indexer score: iscore[t,s] = sum_h relu(iq[t,h,:]·ik[s,:])*c*iw[t,h]
// WMMA over K=128 per head; causal mask (-inf) applied on store.
// Tile: 128 (t) x 64 (s); 8 waves 4x2, each wave 32x32 (2x2 WMMA tiles).
// ---------------------------------------------------------------------------
__global__ __launch_bounds__(256) void k_iscore(const float* __restrict__ iq,
                                                const float* __restrict__ ik,
                                                const float* __restrict__ iw,
                                                float* __restrict__ iscore)
{
  constexpr int BMr=128, BNs=64, BK=32, LS=40;
  const int tid=threadIdx.x, wave=tid>>5, lane=tid&31;
  const int wM=wave&3, wN=wave>>2;
  const int half=lane>>4, lr=lane&15;
  const int tT=blockIdx.y*BMr, tS=blockIdx.x*BNs;
  if (tS > tT + (BMr-1)) { // fully above diagonal
    for (int i=tid;i<BMr*BNs;i+=256){
      int r=i>>6, c=i&63;
      iscore[(size_t)(tT+r)*T_ + tS + c] = -__builtin_inff();
    }
    return;
  }
  __shared__ __align__(16) __bf16 sA[BMr*LS];
  __shared__ __align__(16) __bf16 sB[BNs*LS];
  __shared__ float siw[BMr];

  const float cscale = 0.08838834764831845f   /* ID^-0.5 */
                     * 0.17677669529663687f;  /* IH^-0.5 */
  const f32x8 z8 = {0.f,0.f,0.f,0.f,0.f,0.f,0.f,0.f};
  float racc[2][2][8];
#pragma unroll
  for (int a=0;a<2;++a)
#pragma unroll
    for (int b=0;b<2;++b)
#pragma unroll
      for (int j=0;j<8;++j) racc[a][b][j]=0.f;

  for (int hh=0; hh<IH_; ++hh) {
    __syncthreads();                    // protect siw reuse across heads
    if (tid < BMr) siw[tid] = iw[(size_t)(tT+tid)*IH_ + hh];
    f32x8 sacc[2][2];
#pragma unroll
    for (int a=0;a<2;++a)
#pragma unroll
      for (int b=0;b<2;++b) sacc[a][b]=z8;

    for (int k0=0;k0<ID_;k0+=BK){
#pragma unroll
      for (int i=tid;i<(BMr*BK)/4;i+=256){
        int r=i>>3, c=(i&7)*4;
        float4 v = *(const float4*)&iq[(size_t)(tT+r)*(IH_*ID_) + hh*ID_ + k0 + c];
        *(bf16x4*)&sA[r*LS+c] = cvt4(v);
      }
#pragma unroll
      for (int i=tid;i<(BNs*BK)/4;i+=256){
        int n=i>>3, c=(i&7)*4;
        float4 v = *(const float4*)&ik[(size_t)(tS+n)*ID_ + k0 + c];
        *(bf16x4*)&sB[n*LS+c] = cvt4(v);
      }
      __syncthreads();
      Frag a[2], b[2];
#pragma unroll
      for (int mt=0;mt<2;++mt){
        int r = wM*32 + mt*16 + lr;
        a[mt].h[0] = *(const bf16x8*)&sA[r*LS + half*8];
        a[mt].h[1] = *(const bf16x8*)&sA[r*LS + 16 + half*8];
      }
#pragma unroll
      for (int nt=0;nt<2;++nt){
        int c = wN*32 + nt*16 + lr;
        b[nt].h[0] = *(const bf16x8*)&sB[c*LS + half*8];
        b[nt].h[1] = *(const bf16x8*)&sB[c*LS + 16 + half*8];
      }
#pragma unroll
      for (int mt=0;mt<2;++mt)
#pragma unroll
        for (int nt=0;nt<2;++nt)
          sacc[mt][nt] = wmma_bf16(a[mt], b[nt], sacc[mt][nt]);
      __syncthreads();
    }
#pragma unroll
    for (int mt=0;mt<2;++mt)
#pragma unroll
      for (int nt=0;nt<2;++nt)
#pragma unroll
        for (int j=0;j<8;++j){
          int rl = wM*32 + mt*16 + half*8 + j;
          racc[mt][nt][j] += fmaxf(sacc[mt][nt][j],0.f)*cscale*siw[rl];
        }
  }
#pragma unroll
  for (int mt=0;mt<2;++mt)
#pragma unroll
    for (int nt=0;nt<2;++nt)
#pragma unroll
      for (int j=0;j<8;++j){
        int row = tT + wM*32 + mt*16 + half*8 + j;
        int col = tS + wN*32 + nt*16 + lr;
        float v = (col<=row)? racc[mt][nt][j] : -__builtin_inff();
        iscore[(size_t)row*T_ + col] = v;
      }
}

// ---------------------------------------------------------------------------
// Top-512 per row via MSB-first radix select on sign-flipped float keys.
// ---------------------------------------------------------------------------
__global__ __launch_bounds__(256) void k_topk(const float* __restrict__ iscore,
                                              unsigned char* __restrict__ sel)
{
  const int t=blockIdx.x, tid=threadIdx.x;
  const float* row = iscore + (size_t)t*T_;
  unsigned char* sr = sel + (size_t)t*T_;
  const int n = t+1;
  if (n <= TOPK_) {
    for (int s=tid;s<T_;s+=256) sr[s] = (s<=t)?1:0;
    return;
  }
  __shared__ int hist[256];
  unsigned prefix=0; int remaining=TOPK_;
  for (int pass=3; pass>=0; --pass) {
    hist[tid]=0;
    __syncthreads();
    const int sh = pass*8;
    const unsigned hiMask = (pass==3)? 0u : (0xFFFFFFFFu << (sh+8));
    for (int s=tid;s<n;s+=256){
      unsigned u=fkey(row[s]);
      if ((u&hiMask)==(prefix&hiMask)) atomicAdd(&hist[(u>>sh)&255],1);
    }
    __syncthreads();
    int cum=0, b=0;
    for (int bb=255;bb>=0;--bb){
      int c=hist[bb];
      if (cum + c >= remaining){ b=bb; break; }
      cum+=c;
    }
    remaining -= cum;
    prefix |= ((unsigned)b) << sh;
    __syncthreads();
  }
  for (int s=tid;s<T_;s+=256)
    sr[s] = (s<=t && fkey(row[s])>=prefix) ? 1 : 0;
}

// ---------------------------------------------------------------------------
__global__ __launch_bounds__(256) void k_rmsnorm(const float* __restrict__ x, int ldx,
                                                 const float* __restrict__ w,
                                                 float* __restrict__ y, int ldy, int cols)
{
  const int t=blockIdx.x, tid=threadIdx.x;
  const float* xr = x + (size_t)t*ldx;
  float* yr = y + (size_t)t*ldy;
  __shared__ float red[256];
  float ss=0.f;
  for (int c=tid;c<cols;c+=256){ float v=xr[c]; ss+=v*v; }
  red[tid]=ss; __syncthreads();
  for (int st=128;st>0;st>>=1){ if(tid<st) red[tid]+=red[tid+st]; __syncthreads(); }
  float rs = rsqrtf(red[0]/(float)cols + 1e-6f);
  for (int c=tid;c<cols;c+=256) yr[c]=xr[c]*rs*w[c];
}

__global__ __launch_bounds__(128) void k_ln128(float* __restrict__ x,
                                               const float* __restrict__ w,
                                               const float* __restrict__ b)
{
  const int t=blockIdx.x, tid=threadIdx.x;
  float* xr = x + (size_t)t*ID_;
  __shared__ float red[128];
  float v = xr[tid];
  red[tid]=v; __syncthreads();
  for (int st=64;st>0;st>>=1){ if(tid<st) red[tid]+=red[tid+st]; __syncthreads(); }
  float mean = red[0]*(1.f/128.f); __syncthreads();
  float d = v-mean;
  red[tid]=d*d; __syncthreads();
  for (int st=64;st>0;st>>=1){ if(tid<st) red[tid]+=red[tid+st]; __syncthreads(); }
  float var = red[0]*(1.f/128.f);
  xr[tid] = d*rsqrtf(var+1e-6f)*w[tid]+b[tid];
}

__global__ __launch_bounds__(256) void k_softmax_sel(float* __restrict__ sc,
                                                     const unsigned char* __restrict__ sel,
                                                     float scale)
{
  const int t=blockIdx.x, tid=threadIdx.x;
  float* row = sc + (size_t)t*T_;
  const unsigned char* sr = sel + (size_t)t*T_;
  __shared__ float red[256];
  float m=-__builtin_inff();
  for (int s=tid;s<T_;s+=256) if (sr[s]) m=fmaxf(m,row[s]*scale);
  red[tid]=m; __syncthreads();
  for (int st=128;st>0;st>>=1){ if(tid<st) red[tid]=fmaxf(red[tid],red[tid+st]); __syncthreads(); }
  m = red[0]; __syncthreads();
  float sum=0.f;
  for (int s=tid;s<T_;s+=256){
    float e = sr[s]? __expf(row[s]*scale - m) : 0.f;
    row[s]=e; sum+=e;
  }
  red[tid]=sum; __syncthreads();
  for (int st=128;st>0;st>>=1){ if(tid<st) red[tid]+=red[tid+st]; __syncthreads(); }
  float inv = 1.f/red[0];
  for (int s=tid;s<T_;s+=256) row[s]*=inv;
}

// interleaved RoPE on q_pe (in place): q is (T, H, 192), rope on dims [128,192)
__global__ __launch_bounds__(256) void k_rope_q(float* __restrict__ q,
                                                const int* __restrict__ positions)
{
  int t=blockIdx.x; float pos=(float)positions[t];
  for (int p=threadIdx.x; p<H_*32; p+=256){
    int hh=p>>5, j=p&31;
    size_t base=(size_t)t*(H_*QKD_) + hh*QKD_ + NOPE_ + 2*j;
    float ang=pos*rope_freq(j);
    float c=__cosf(ang), s=__sinf(ang);
    float x1=q[base], x2=q[base+1];
    q[base]   = x1*c - x2*s;
    q[base+1] = x2*c + x1*s;
  }
}

// kf (T,H,192) = [k_nope from kv | roped k_pe broadcast]
__global__ __launch_bounds__(256) void k_build_kf(const float* __restrict__ kv,
                                                  const float* __restrict__ qkv,
                                                  const int* __restrict__ positions,
                                                  float* __restrict__ kf)
{
  int t=blockIdx.x; float pos=(float)positions[t];
  for (int i=threadIdx.x; i<H_*QKD_; i+=256){
    int hh=i/QKD_, d=i-hh*QKD_;
    float v;
    if (d<NOPE_) {
      v = kv[(size_t)t*(H_*(NOPE_+VD_)) + hh*(NOPE_+VD_) + d];
    } else {
      int dd=d-NOPE_, j=dd>>1;
      float ang=pos*rope_freq(j);
      float c=__cosf(ang), s=__sinf(ang);
      float x1=qkv[(size_t)t*QKV_N + QL_+KVL_ + 2*j];
      float x2=qkv[(size_t)t*QKV_N + QL_+KVL_ + 2*j+1];
      v = (dd&1)? (x2*c+x1*s) : (x1*c-x2*s);
    }
    kf[(size_t)t*(H_*QKD_)+i]=v;
  }
}

// NeoX RoPE on first 64 dims of each 128-dim index head (in place)
__global__ __launch_bounds__(256) void k_rope_iq(float* __restrict__ iq,
                                                 const int* __restrict__ positions)
{
  int t=blockIdx.x; float pos=(float)positions[t];
  for (int p=threadIdx.x; p<IH_*32; p+=256){
    int hh=p>>5, j=p&31;
    size_t base=(size_t)t*(IH_*ID_) + hh*ID_;
    float ang=pos*rope_freq(j);
    float c=__cosf(ang), s=__sinf(ang);
    float x1=iq[base+j], x2=iq[base+32+j];
    iq[base+j]    = x1*c - x2*s;
    iq[base+32+j] = x2*c + x1*s;
  }
}

__global__ __launch_bounds__(32) void k_rope_ik(float* __restrict__ ik,
                                                const int* __restrict__ positions)
{
  int t=blockIdx.x, j=threadIdx.x;
  float pos=(float)positions[t];
  float ang=pos*rope_freq(j);
  float c=__cosf(ang), s=__sinf(ang);
  size_t base=(size_t)t*ID_;
  float x1=ik[base+j], x2=ik[base+32+j];
  ik[base+j]    = x1*c - x2*s;
  ik[base+32+j] = x2*c + x1*s;
}

// routing: sigmoid gate, top-4 on (gsc+bias), weights from gsc, renorm * 2.5
__global__ __launch_bounds__(256) void k_gate_topk(const float* __restrict__ logits,
                                                   const float* __restrict__ bias,
                                                   int* __restrict__ eidx,
                                                   float* __restrict__ gwt)
{
  int t = blockIdx.x*256 + threadIdx.x;
  if (t>=T_) return;
  float gsc[E_], sc[E_];
  for (int e=0;e<E_;++e){ float g=sigmoidf_(logits[t*E_+e]); gsc[e]=g; sc[e]=g+bias[e]; }
  float sum=0.f; int pick[TK_];
  for (int k=0;k<TK_;++k){
    float best=-__builtin_inff(); int bi=0;
    for (int e=0;e<E_;++e) if (sc[e]>best){best=sc[e];bi=e;}
    pick[k]=bi; sc[bi]=-__builtin_inff(); sum+=gsc[bi];
  }
  float inv = 2.5f/(sum+1e-20f);
  for (int k=0;k<TK_;++k){ eidx[t*TK_+k]=pick[k]; gwt[t*TK_+k]=gsc[pick[k]]*inv; }
}

__global__ void k_zero_i32(int* __restrict__ p, int n){
  int i=blockIdx.x*blockDim.x+threadIdx.x; if(i<n)p[i]=0;
}
__global__ void k_count(const int* __restrict__ eidx, int* __restrict__ cnt){
  int i=blockIdx.x*blockDim.x+threadIdx.x;
  if (i<T_*TK_) atomicAdd(&cnt[eidx[i]],1);
}
__global__ void k_scan(const int* __restrict__ cnt, int* __restrict__ seg, int* __restrict__ cur){
  if (threadIdx.x==0 && blockIdx.x==0){
    int s=0;
    for (int e=0;e<E_;++e){ seg[e]=s; cur[e]=s; s+=cnt[e]; }
    seg[E_]=s;
  }
}
__global__ void k_scatter(const int* __restrict__ eidx, const float* __restrict__ gwt,
                          int* __restrict__ cur, int* __restrict__ tokmap,
                          float* __restrict__ gwperm){
  int i=blockIdx.x*blockDim.x+threadIdx.x;
  if (i>=T_*TK_) return;
  int e=eidx[i];
  int pos=atomicAdd(&cur[e],1);
  tokmap[pos]=i>>2;          // token index
  gwperm[pos]=gwt[i];
}

__global__ void k_add(const float* __restrict__ a, const float* __restrict__ b,
                      float* __restrict__ c, int n){
  int i=blockIdx.x*blockDim.x+threadIdx.x; if(i<n)c[i]=a[i]+b[i];
}
__global__ void k_silu_mul(const float* __restrict__ g, const float* __restrict__ u,
                           float* __restrict__ o, int n){
  int i=blockIdx.x*blockDim.x+threadIdx.x; if(i<n)o[i]=siluf_(g[i])*u[i];
}
__global__ void k_silu_mul_gw(const float* __restrict__ g, const float* __restrict__ u,
                              const float* __restrict__ gw, float* __restrict__ o,
                              int n, int cols){
  int i=blockIdx.x*blockDim.x+threadIdx.x;
  if (i<n){ int r=i/cols; o[i]=siluf_(g[i])*u[i]*gw[r]; }
}

} // namespace

// ---------------------------------------------------------------------------
extern "C" void kernel_launch(void* const* d_in, const int* in_sizes, int n_in,
                              void* d_out, int out_size, void* d_ws, size_t ws_size,
                              hipStream_t stream)
{
  (void)in_sizes; (void)n_in; (void)out_size; (void)ws_size;

  const int*   positions = (const int*)  d_in[0];
  const float* hidden    = (const float*)d_in[1];
  const float* w_in_ln   = (const float*)d_in[2];
  const float* w_post_ln = (const float*)d_in[3];
  const float* W_qkv_a   = (const float*)d_in[4];
  const float* q_a_ln_w  = (const float*)d_in[5];
  const float* kv_a_ln_w = (const float*)d_in[6];
  const float* W_q_b     = (const float*)d_in[7];
  const float* W_kv_b    = (const float*)d_in[8];
  const float* W_o       = (const float*)d_in[9];
  const float* W_iq_b    = (const float*)d_in[10];
  const float* W_ik      = (const float*)d_in[11];
  const float* idx_ln_w  = (const float*)d_in[12];
  const float* idx_ln_b  = (const float*)d_in[13];
  const float* W_iw      = (const float*)d_in[14];
  const float* W_gate    = (const float*)d_in[15];
  const float* gate_bias = (const float*)d_in[16];
  const float* We_gate   = (const float*)d_in[17];
  const float* We_up     = (const float*)d_in[18];
  const float* We_down   = (const float*)d_in[19];
  const float* Ws_gate   = (const float*)d_in[20];
  const float* Ws_up     = (const float*)d_in[21];
  const float* Ws_down   = (const float*)d_in[22];

  float* out_y   = (float*)d_out;                     // y + shared  (T x D)
  float* out_res = (float*)d_out + (size_t)T_*D_;     // res2        (T x D)

  char* wsp = (char*)d_ws;
  size_t woff = 0;
  auto alloc = [&](size_t bytes)->void*{
    void* p = wsp + woff;
    woff += (bytes + 255) & ~(size_t)255;
    return p;
  };

  float* h       = (float*)alloc(sizeof(float)*(size_t)T_*D_);
  float* qkv     = (float*)alloc(sizeof(float)*(size_t)T_*QKV_N);
  float* q_cn    = (float*)alloc(sizeof(float)*(size_t)T_*QL_);
  float* kv_cn   = (float*)alloc(sizeof(float)*(size_t)T_*KVL_);
  float* q       = (float*)alloc(sizeof(float)*(size_t)T_*H_*QKD_);
  float* kv      = (float*)alloc(sizeof(float)*(size_t)T_*H_*(NOPE_+VD_));
  float* kf      = (float*)alloc(sizeof(float)*(size_t)T_*H_*QKD_);
  float* iq      = (float*)alloc(sizeof(float)*(size_t)T_*IH_*ID_);
  float* ik      = (float*)alloc(sizeof(float)*(size_t)T_*ID_);
  float* iw      = (float*)alloc(sizeof(float)*(size_t)T_*IH_);
  float* iscore  = (float*)alloc(sizeof(float)*(size_t)T_*T_);
  unsigned char* sel = (unsigned char*)alloc((size_t)T_*T_);
  float* scores  = (float*)alloc(sizeof(float)*(size_t)T_*T_);
  float* attn    = (float*)alloc(sizeof(float)*(size_t)T_*H_*VD_);
  float* oproj   = (float*)alloc(sizeof(float)*(size_t)T_*D_);
  float* h2      = (float*)alloc(sizeof(float)*(size_t)T_*D_);
  float* glog    = (float*)alloc(sizeof(float)*(size_t)T_*E_);
  int*   eidx    = (int*)  alloc(sizeof(int)*(size_t)T_*TK_);
  float* gwt     = (float*)alloc(sizeof(float)*(size_t)T_*TK_);
  int*   cnt     = (int*)  alloc(sizeof(int)*E_);
  int*   seg     = (int*)  alloc(sizeof(int)*(E_+1));
  int*   cur     = (int*)  alloc(sizeof(int)*E_);
  int*   tokmap  = (int*)  alloc(sizeof(int)*(size_t)T_*TK_);
  float* gwperm  = (float*)alloc(sizeof(float)*(size_t)T_*TK_);
  float* g_sh    = (float*)alloc(sizeof(float)*(size_t)T_*ISH_);
  float* u_sh    = (float*)alloc(sizeof(float)*(size_t)T_*ISH_);
  float* gateE   = (float*)alloc(sizeof(float)*(size_t)T_*TK_*I_);
  float* upE     = (float*)alloc(sizeof(float)*(size_t)T_*TK_*I_);

  auto gemm = [&](const float* A,const float* B,float* C,int M,int N,int K,
                  int lda,int ldb,int ldc){
    dim3 g((N+127)/128,(M+127)/128);
    k_gemm<false><<<g,256,0,stream>>>(A,B,C,M,N,K,lda,ldb,ldc);
  };
  auto gemmBT = [&](const float* A,const float* B,float* C,int M,int N,int K,
                    int lda,int ldb,int ldc){
    dim3 g((N+127)/128,(M+127)/128);
    k_gemm<true><<<g,256,0,stream>>>(A,B,C,M,N,K,lda,ldb,ldc);
  };

  // ---- attention input path ----
  k_rmsnorm<<<T_,256,0,stream>>>(hidden, D_, w_in_ln, h, D_, D_);
  gemm(h, W_qkv_a, qkv, T_, QKV_N, D_, D_, QKV_N, QKV_N);
  k_rmsnorm<<<T_,256,0,stream>>>(qkv,        QKV_N, q_a_ln_w,  q_cn,  QL_,  QL_);
  k_rmsnorm<<<T_,256,0,stream>>>(qkv + QL_,  QKV_N, kv_a_ln_w, kv_cn, KVL_, KVL_);
  gemm(q_cn,  W_q_b,  q,  T_, H_*QKD_,        QL_,  QL_,  H_*QKD_,        H_*QKD_);
  gemm(kv_cn, W_kv_b, kv, T_, H_*(NOPE_+VD_), KVL_, KVL_, H_*(NOPE_+VD_), H_*(NOPE_+VD_));
  gemm(q_cn,  W_iq_b, iq, T_, IH_*ID_,        QL_,  QL_,  IH_*ID_,        IH_*ID_);
  gemm(h,     W_ik,   ik, T_, ID_,            D_,   D_,   ID_,            ID_);
  gemm(h,     W_iw,   iw, T_, IH_,            D_,   D_,   IH_,            IH_);

  k_rope_q<<<T_,256,0,stream>>>(q, positions);           // q -> qf (in place)
  k_build_kf<<<T_,256,0,stream>>>(kv, qkv, positions, kf);
  k_rope_iq<<<T_,256,0,stream>>>(iq, positions);
  k_ln128<<<T_,128,0,stream>>>(ik, idx_ln_w, idx_ln_b);
  k_rope_ik<<<T_,32,0,stream>>>(ik, positions);

  // ---- indexer scores + top-512 selection ----
  {
    dim3 gi(T_/64, T_/128);
    k_iscore<<<gi,256,0,stream>>>(iq, ik, iw, iscore);
  }
  k_topk<<<T_,256,0,stream>>>(iscore, sel);

  // ---- sparse attention, per head ----
  const float att_scale = 0.07216878364870323f; // QKD^-0.5
  for (int hh=0; hh<H_; ++hh) {
    gemmBT(q  + hh*QKD_, kf + hh*QKD_, scores, T_, T_, QKD_, H_*QKD_, H_*QKD_, T_);
    k_softmax_sel<<<T_,256,0,stream>>>(scores, sel, att_scale);
    gemm(scores, kv + hh*(NOPE_+VD_) + NOPE_, attn + hh*VD_,
         T_, VD_, T_, T_, H_*(NOPE_+VD_), H_*VD_);
  }
  gemm(attn, W_o, oproj, T_, D_, H_*VD_, H_*VD_, D_, D_);

  // ---- residual + post-norm ----
  k_add<<<(T_*D_+255)/256,256,0,stream>>>(hidden, oproj, out_res, T_*D_);
  k_rmsnorm<<<T_,256,0,stream>>>(out_res, D_, w_post_ln, h2, D_, D_);

  // ---- MoE routing ----
  gemm(h2, W_gate, glog, T_, E_, D_, D_, E_, E_);
  k_gate_topk<<<(T_+255)/256,256,0,stream>>>(glog, gate_bias, eidx, gwt);
  k_zero_i32<<<1,64,0,stream>>>(cnt, E_);
  k_count<<<(T_*TK_+255)/256,256,0,stream>>>(eidx, cnt);
  k_scan<<<1,1,0,stream>>>(cnt, seg, cur);
  k_scatter<<<(T_*TK_+255)/256,256,0,stream>>>(eidx, gwt, cur, tokmap, gwperm);

  // ---- shared expert -> out_y (plain store, fully initializes out_y) ----
  gemm(h2, Ws_gate, g_sh, T_, ISH_, D_, D_, ISH_, ISH_);
  gemm(h2, Ws_up,   u_sh, T_, ISH_, D_, D_, ISH_, ISH_);
  k_silu_mul<<<(T_*ISH_+255)/256,256,0,stream>>>(g_sh, u_sh, g_sh, T_*ISH_);
  gemm(g_sh, Ws_down, out_y, T_, D_, ISH_, ISH_, D_, D_);

  // ---- routed experts (gathered grouped GEMMs), scatter-add into out_y ----
  {
    dim3 gg((I_+127)/128, (T_*TK_)/128, E_);
    k_gemm_grp<true,false><<<gg,256,0,stream>>>(h2, We_gate, gateE, tokmap, seg,
                                                I_, D_, D_, (long long)D_*I_, I_);
    k_gemm_grp<true,false><<<gg,256,0,stream>>>(h2, We_up,   upE,   tokmap, seg,
                                                I_, D_, D_, (long long)D_*I_, I_);
  }
  k_silu_mul_gw<<<((T_*TK_*I_)+255)/256,256,0,stream>>>(gateE, upE, gwperm, gateE,
                                                        T_*TK_*I_, I_);
  {
    dim3 gd((D_+127)/128, (T_*TK_)/128, E_);
    k_gemm_grp<false,true><<<gd,256,0,stream>>>(gateE, We_down, out_y, tokmap, seg,
                                                D_, I_, I_, (long long)I_*D_, D_);
  }
}
